// cross_attention_8624294330992
// MI455X (gfx1250) — matrix operand order repaired and checked by reading, not compiled, not run on hardware
//
#include <hip/hip_runtime.h>
#include <math.h>
#include <stdint.h>

// ---------------------------------------------------------------------------
// Types for CDNA5 WMMA (wave32): v_wmma_f32_16x16x32_bf16
// ---------------------------------------------------------------------------
typedef __attribute__((ext_vector_type(8)))  __bf16 v8bf;
typedef __attribute__((ext_vector_type(16))) __bf16 v16bf;
typedef __attribute__((ext_vector_type(8)))  float  v8f;
typedef __attribute__((ext_vector_type(4)))  int    v4i;

// Native bf16 conversion (RNE) -> hardware cvt on gfx1250
static __device__ __forceinline__ __bf16 f2bf(float f) { return (__bf16)f; }

static __device__ __forceinline__ v16bf cat16(v8bf lo, v8bf hi) {
  return __builtin_shufflevector(lo, hi, 0,1,2,3,4,5,6,7,8,9,10,11,12,13,14,15);
}

// B-fragment (32x16): lane holds col n = lane&15, K = 16*half + i  -> 16
// contiguous bf16 when the matrix is stored K-contiguous per column.
static __device__ __forceinline__ v16bf ldB(const __bf16* p) {
  v8bf lo = *(const v8bf*)p;
  v8bf hi = *(const v8bf*)(p + 8);
  return cat16(lo, hi);
}

// A-fragment (16x32): lane holds row m = lane&15; K = 16h+[0..15]
static __device__ __forceinline__ v16bf ldA(const __bf16* rowp, int h) {
  v8bf lo = *(const v8bf*)(rowp + 16 * h);
  v8bf hi = *(const v8bf*)(rowp + 16 * h + 8);
  return cat16(lo, hi);
}

static __device__ __forceinline__ v8f wmma_bf16(v16bf a, v16bf b, v8f c) {
  return __builtin_amdgcn_wmma_f32_16x16x32_bf16(false, a, false, b, (short)0, c,
                                                 false, false);
}

// ---------------------------------------------------------------------------
// Async global->LDS (CDNA5 GLOBAL_LOAD_ASYNC_TO_LDS_B128, ASYNCcnt-tracked).
// Toolchain signature (from diagnostics): (int4* src, int4* lds, imm, imm)
// with generic pointers; one int4 (16B) moved per lane.
// ---------------------------------------------------------------------------
#if defined(__HIP_DEVICE_COMPILE__) && \
    __has_builtin(__builtin_amdgcn_global_load_async_to_lds_b128)
#define USE_ASYNC 1
#else
#define USE_ASYNC 0
#endif

static __device__ __forceinline__ void cp16_g2l(void* l, const void* g) {
#if USE_ASYNC
  __builtin_amdgcn_global_load_async_to_lds_b128((v4i*)g, (v4i*)l, 0, 0);
#else
  *(v8bf*)l = *(const v8bf*)g;
#endif
}

static __device__ __forceinline__ void async_wait_all() {
#if USE_ASYNC
#if __has_builtin(__builtin_amdgcn_s_wait_asynccnt)
  __builtin_amdgcn_s_wait_asynccnt(0);
#else
  asm volatile("s_wait_asynccnt 0x0" ::: "memory");
#endif
#endif
}

// Reduce across the 16 lanes of each wave-half (C-layout: cols live in lanes)
static __device__ __forceinline__ float rmax16(float v) {
#pragma unroll
  for (int m = 1; m <= 8; m <<= 1) v = fmaxf(v, __shfl_xor(v, m, 32));
  return v;
}
static __device__ __forceinline__ float rsum16(float v) {
#pragma unroll
  for (int m = 1; m <= 8; m <<= 1) v += __shfl_xor(v, m, 32);
  return v;
}

#define NN 8192
#define DD 512

// ---------------------------------------------------------------------------
// Kernel 1: fused projections  Out = bf16( X @ W ),  X fp32 [8192,512],
// W fp32 [512,512].  blockIdx.z selects (x1,q)->Q, (x2,k)->K, (x1,v)->V.
// Block = 256 thr (8 waves), tile = 256 rows x 64 cols, K staged in 32-chunks.
// ---------------------------------------------------------------------------
#define AP 40   // padded LDS row stride (bf16) for A chunk [256][32]
#define BP 40   // padded LDS row stride (bf16) for Bt chunk [64][32]

__global__ __launch_bounds__(256) void proj_kernel(
    const float* __restrict__ x1, const float* __restrict__ x2,
    const float* __restrict__ qw, const float* __restrict__ kw,
    const float* __restrict__ vw,
    __bf16* __restrict__ Qb, __bf16* __restrict__ Kb, __bf16* __restrict__ Vb) {
  extern __shared__ __align__(16) char smem[];
  __bf16* Alds = (__bf16*)smem;        // [256][AP]
  __bf16* Bt   = Alds + 256 * AP;      // [64][BP]  (transposed: Bt[n][k])

  const float* X; const float* W; __bf16* Out;
  if (blockIdx.z == 0)      { X = x1; W = qw; Out = Qb; }
  else if (blockIdx.z == 1) { X = x2; W = kw; Out = Kb; }
  else                      { X = x1; W = vw; Out = Vb; }

  const int n0 = blockIdx.x * 64;
  const int m0 = blockIdx.y * 256;
  const int tid  = threadIdx.x;
  const int w    = tid >> 5;
  const int lane = tid & 31;
  const int h    = lane >> 4;
  const int n15  = lane & 15;

  v8f zero8 = {0.f, 0.f, 0.f, 0.f, 0.f, 0.f, 0.f, 0.f};
  v8f acc[2][4];
#pragma unroll
  for (int rt = 0; rt < 2; ++rt)
#pragma unroll
    for (int jt = 0; jt < 4; ++jt) acc[rt][jt] = zero8;

  for (int k0 = 0; k0 < DD; k0 += 32) {
    __syncthreads();
    // A chunk: 256 rows x 32 k, fp32 -> bf16
    {
      const float4* xp = (const float4*)&X[(size_t)(m0 + tid) * DD + k0];
#pragma unroll
      for (int c = 0; c < 4; ++c) {
        float4 f0 = xp[2 * c];
        float4 f1 = xp[2 * c + 1];
        v8bf bv;
        bv[0] = f2bf(f0.x); bv[1] = f2bf(f0.y); bv[2] = f2bf(f0.z); bv[3] = f2bf(f0.w);
        bv[4] = f2bf(f1.x); bv[5] = f2bf(f1.y); bv[6] = f2bf(f1.z); bv[7] = f2bf(f1.w);
        *(v8bf*)&Alds[tid * AP + 8 * c] = bv;
      }
    }
    // B chunk, stored transposed Bt[n][k]
    for (int i = tid; i < 32 * 64; i += 256) {
      int kk = i >> 6, nn = i & 63;
      Bt[nn * BP + kk] = f2bf(W[(size_t)(k0 + kk) * DD + n0 + nn]);
    }
    __syncthreads();

#pragma unroll
    for (int rt = 0; rt < 2; ++rt) {
      v16bf a = ldA(&Alds[(32 * w + 16 * rt + n15) * AP], h);
#pragma unroll
      for (int jt = 0; jt < 4; ++jt) {
        v16bf b = ldB(&Bt[(16 * jt + n15) * BP + 16 * h]);
        acc[rt][jt] = wmma_bf16(a, b, acc[rt][jt]);
      }
    }
  }

#pragma unroll
  for (int rt = 0; rt < 2; ++rt)
#pragma unroll
    for (int jt = 0; jt < 4; ++jt)
#pragma unroll
      for (int r = 0; r < 8; ++r)
        Out[(size_t)(m0 + 32 * w + 16 * rt + 8 * h + r) * DD + n0 + 16 * jt + n15] =
            f2bf(acc[rt][jt][r]);
}

// ---------------------------------------------------------------------------
// Kernel 2: flash attention.  Block = 256 thr (8 waves) owns 32 query rows.
// Wave (rg,cs): rg = 16-row group, cs = 128-col slice of the output.
// Key blocks of 64; online softmax with leaky_relu(0.01) on logits.
// Q strip and K tiles staged via async global->LDS copies.
// ---------------------------------------------------------------------------
#define QP 520  // LDS stride for Q strip  [32][512]
#define KP 520  // LDS stride for K tile   [64][512]
#define VP 72   // LDS stride for V^T tile [512][64]

__global__ __launch_bounds__(256) void attn_kernel(
    const __bf16* __restrict__ Qb, const __bf16* __restrict__ Kb,
    const __bf16* __restrict__ Vb, float* __restrict__ out) {
  extern __shared__ __align__(16) char smem[];
  __bf16* Qlds = (__bf16*)smem;            // 32*QP
  __bf16* Klds = Qlds + 32 * QP;           // 64*KP
  __bf16* Vt   = Klds + 64 * KP;           // 512*VP  (Vt[d][n])
  __bf16* Plds = Vt + 512 * VP;            // [2][16][64]
  float* redmax = (float*)(Plds + 2 * 16 * 64);  // [2][4][16]
  float* redsum = redmax + 128;                  // [2][4][16]

  const int tid  = threadIdx.x;
  const int wv   = tid >> 5;
  const int lane = tid & 31;
  const int h    = lane >> 4;
  const int n15  = lane & 15;
  const int rg   = wv >> 2;   // row group 0/1
  const int cs   = wv & 3;    // D-slice 0..3 (128 cols each)
  const int m0   = blockIdx.x * 32;

  // Load resident Q strip (32 rows x 512) once, async global->LDS
  for (int i = tid; i < 32 * 64; i += 256) {
    int r = i >> 6, c = (i & 63) * 8;
    cp16_g2l(&Qlds[r * QP + c], &Qb[(size_t)(m0 + r) * DD + c]);
  }

  v8f zero8 = {0.f, 0.f, 0.f, 0.f, 0.f, 0.f, 0.f, 0.f};
  v8f o[8];
#pragma unroll
  for (int jt = 0; jt < 8; ++jt) o[jt] = zero8;
  float Mrow[8], lrow[8];
#pragma unroll
  for (int r = 0; r < 8; ++r) { Mrow[r] = -__builtin_inff(); lrow[r] = 0.f; }

  for (int n0 = 0; n0 < NN; n0 += 64) {
    __syncthreads();
    // Stage K tile (row-major, async) and V tile transposed (explicit)
    for (int i = tid; i < 64 * 64; i += 256) {
      int r = i >> 6, c = (i & 63) * 8;
      cp16_g2l(&Klds[r * KP + c], &Kb[(size_t)(n0 + r) * DD + c]);
    }
    for (int i = tid; i < 64 * 64; i += 256) {
      int n = i >> 6, c = (i & 63) * 8;
      v8bf val = *(const v8bf*)&Vb[(size_t)(n0 + n) * DD + c];
#pragma unroll
      for (int j = 0; j < 8; ++j) Vt[(c + j) * VP + n] = val[j];
    }
    async_wait_all();
    __syncthreads();

    // ---- S = Q . K^T : one 16x16 tile per wave (rows rg, key cols 16*cs) ----
    v8f s = zero8;
    const __bf16* qrow = &Qlds[(16 * rg + n15) * QP];
    const __bf16* krow = &Klds[(16 * cs + n15) * KP];
#pragma unroll
    for (int kk = 0; kk < DD; kk += 32) {
      v16bf a = ldA(qrow + kk, h);
      v16bf b = ldB(krow + kk + 16 * h);
      s = wmma_bf16(a, b, s);
    }

    // leaky_relu + per-row tile max
    float tmax[8];
#pragma unroll
    for (int r = 0; r < 8; ++r) {
      float x = s[r];
      x = (x > 0.f) ? x : 0.01f * x;
      s[r] = x;
      tmax[r] = rmax16(x);
    }
    if (n15 == 0) {
#pragma unroll
      for (int r = 0; r < 8; ++r) redmax[(rg * 4 + cs) * 16 + 8 * h + r] = tmax[r];
    }
    __syncthreads();

    // combine max across the 4 column-tiles of this row group
    float scale[8];
#pragma unroll
    for (int r = 0; r < 8; ++r) {
      int row = 8 * h + r;
      float mb = fmaxf(fmaxf(redmax[(rg * 4 + 0) * 16 + row],
                             redmax[(rg * 4 + 1) * 16 + row]),
                       fmaxf(redmax[(rg * 4 + 2) * 16 + row],
                             redmax[(rg * 4 + 3) * 16 + row]));
      float Mn = fmaxf(Mrow[r], mb);
      scale[r] = __expf(Mrow[r] - Mn);
      Mrow[r]  = Mn;
    }

    // P = exp(S - Mnew); stash bf16 P tile; partial row sums
    float tsum[8];
#pragma unroll
    for (int r = 0; r < 8; ++r) {
      float p = __expf(s[r] - Mrow[r]);
      tsum[r] = rsum16(p);
      Plds[rg * 1024 + (8 * h + r) * 64 + 16 * cs + n15] = f2bf(p);
      lrow[r] *= scale[r];
    }
    if (n15 == 0) {
#pragma unroll
      for (int r = 0; r < 8; ++r) redsum[(rg * 4 + cs) * 16 + 8 * h + r] = tsum[r];
    }
#pragma unroll
    for (int jt = 0; jt < 8; ++jt)
#pragma unroll
      for (int r = 0; r < 8; ++r) o[jt][r] *= scale[r];
    __syncthreads();

#pragma unroll
    for (int r = 0; r < 8; ++r) {
      int row = 8 * h + r;
      lrow[r] += redsum[(rg * 4 + 0) * 16 + row] + redsum[(rg * 4 + 1) * 16 + row] +
                 redsum[(rg * 4 + 2) * 16 + row] + redsum[(rg * 4 + 3) * 16 + row];
    }

    // ---- O += P . V  (wave's 128-wide D slice) ----
    const __bf16* prow = &Plds[rg * 1024 + n15 * 64];
#pragma unroll
    for (int ns = 0; ns < 2; ++ns) {
      v16bf a = ldA(prow + 32 * ns, h);
#pragma unroll
      for (int jt = 0; jt < 8; ++jt) {
        v16bf b = ldB(&Vt[(128 * cs + 16 * jt + n15) * VP + 32 * ns + 16 * h]);
        o[jt] = wmma_bf16(a, b, o[jt]);
      }
    }
  }

  // normalize and write fp32 output
  float inv[8];
#pragma unroll
  for (int r = 0; r < 8; ++r) inv[r] = 1.0f / lrow[r];
#pragma unroll
  for (int jt = 0; jt < 8; ++jt)
#pragma unroll
    for (int r = 0; r < 8; ++r)
      out[(size_t)(m0 + 16 * rg + 8 * h + r) * DD + 128 * cs + 16 * jt + n15] =
          o[jt][r] * inv[r];
}

// ---------------------------------------------------------------------------
extern "C" void kernel_launch(void* const* d_in, const int* in_sizes, int n_in,
                              void* d_out, int out_size, void* d_ws, size_t ws_size,
                              hipStream_t stream) {
  (void)in_sizes; (void)n_in; (void)out_size; (void)ws_size;
  const float* x1 = (const float*)d_in[0];
  const float* x2 = (const float*)d_in[1];
  const float* qw = (const float*)d_in[2];
  const float* kw = (const float*)d_in[3];
  const float* vw = (const float*)d_in[4];
  float* out = (float*)d_out;

  const size_t MAT_BYTES = (size_t)NN * DD * 2;  // bf16 matrix
  char* wsb = (char*)d_ws;
  __bf16* Qb = (__bf16*)(wsb);
  __bf16* Kb = (__bf16*)(wsb + MAT_BYTES);
  __bf16* Vb = (__bf16*)(wsb + 2 * MAT_BYTES);

  const int PROJ_LDS = (256 * AP + 64 * BP) * 2;                                // 25600 B
  const int ATTN_LDS = (32 * QP + 64 * KP + 512 * VP + 2 * 16 * 64) * 2 + 1024; // 178688 B

  proj_kernel<<<dim3(8, 32, 3), 256, PROJ_LDS, stream>>>(x1, x2, qw, kw, vw,
                                                         Qb, Kb, Vb);
  attn_kernel<<<dim3(NN / 32), 256, ATTN_LDS, stream>>>(Qb, Kb, Vb, out);
}
